// RnnGnnAgentBase_42210938585346
// MI455X (gfx1250) — matrix operand order repaired and checked by reading, not compiled, not run on hardware
//
#include <hip/hip_runtime.h>
#include <hip/hip_bf16.h>

typedef __attribute__((ext_vector_type(16))) _Float16 v16h;
typedef __attribute__((ext_vector_type(8)))  _Float16 v8h;
typedef __attribute__((ext_vector_type(8)))  float    v8f;
typedef __attribute__((ext_vector_type(4)))  float    v4f;

#define N_NODES 50000
#define N_EDGES 800000
#define IN_F    96
#define H_F     128
#define M_F     128
#define G_F     64

// f16 weight workspace offsets (in halfs)
#define OFF_W0   0
#define OFF_W1   12288          // 128*96
#define OFF_WIH  28672          // + 128*128
#define OFF_WHH  77824          // + 384*128
#define OFF_WL   126976         // + 384*128
#define OFF_WR   135168         // + 64*128
#define WH_TOTAL 143360         // + 64*128

// ---------------------------------------------------------------- helpers

__device__ __forceinline__ v8f wmma_f16(v16h a, v16h b, v8f c) {
    // 8 args: (neg_a, A, neg_b, B, c_mod, C, reuse_a, reuse_b)
    return __builtin_amdgcn_wmma_f32_16x16x32_f16(false, a, false, b,
                                                  (short)0, c, false, false);
}

// A fragment (16x32 f16) loaded from a row-major f32 matrix row.
// lane: M = lane&15, half = lane>>4.  K chunks: k0+half*8..+7, k0+16+half*8..+7
__device__ __forceinline__ v16h loadA_g(const float* __restrict__ row, int k0, int half) {
    v16h a;
    const float* p0 = row + k0 + half * 8;
    const float* p1 = p0 + 16;
#pragma unroll
    for (int i = 0; i < 8; ++i) {
        a[i]     = (_Float16)p0[i];
        a[8 + i] = (_Float16)p1[i];
    }
    return a;
}

// A fragment from an LDS f16 buffer [16][ld]
__device__ __forceinline__ v16h loadA_lds(const _Float16* buf, int ld, int k0, int m, int half) {
    const _Float16* p = buf + m * ld + k0 + half * 8;
    v8h lo = *(const v8h*)p;
    v8h hi = *(const v8h*)(p + 16);
    return __builtin_shufflevector(lo, hi, 0,1,2,3,4,5,6,7,8,9,10,11,12,13,14,15);
}

// B fragment (32x16 f16) from f16 weight matrix W[kout][kin] (B col n = W row obase+n).
// lane: N = lane&15, half = lane>>4. K chunks: k0+half*8..+7, k0+16+half*8..+7
__device__ __forceinline__ v16h loadB(const _Float16* __restrict__ W, int kin,
                                      int obase, int k0, int n, int half) {
    const _Float16* p = W + (size_t)(obase + n) * kin + k0 + half * 8;
    v8h lo = *(const v8h*)p;
    v8h hi = *(const v8h*)(p + 16);
    return __builtin_shufflevector(lo, hi, 0,1,2,3,4,5,6,7,8,9,10,11,12,13,14,15);
}

// order-preserving f32 <-> u32 encoding for atomic segment-max
__device__ __forceinline__ unsigned enc_f32(float f) {
    unsigned u = __float_as_uint(f);
    return (u & 0x80000000u) ? ~u : (u | 0x80000000u);
}
__device__ __forceinline__ float dec_f32(unsigned e) {
    unsigned u = (e & 0x80000000u) ? (e & 0x7fffffffu) : ~e;
    return __uint_as_float(u);
}
#define ENC_NEG_INF 0x007FFFFFu   // enc(-inf)

__device__ __forceinline__ void fadd_agent(float* p, float v) {
    __hip_atomic_fetch_add(p, v, __ATOMIC_RELAXED, __HIP_MEMORY_SCOPE_AGENT);
}

__device__ __forceinline__ float sigmoidf_(float x) { return 1.f / (1.f + __expf(-x)); }
__device__ __forceinline__ float tanhf_(float x) {
    float e2 = __expf(2.f * x);
    return 1.f - 2.f / (e2 + 1.f);
}

// ---------------------------------------------------------------- kernels

__global__ void cvt_weights(const float* __restrict__ W0, const float* __restrict__ W1,
                            const float* __restrict__ Wih, const float* __restrict__ Whh,
                            const float* __restrict__ Wl, const float* __restrict__ Wr,
                            _Float16* __restrict__ wh) {
    int i = blockIdx.x * blockDim.x + threadIdx.x;
    if (i >= WH_TOTAL) return;
    float v;
    if      (i < OFF_W1)   v = W0[i - OFF_W0];
    else if (i < OFF_WIH)  v = W1[i - OFF_W1];
    else if (i < OFF_WHH)  v = Wih[i - OFF_WIH];
    else if (i < OFF_WL)   v = Whh[i - OFF_WHH];
    else if (i < OFF_WR)   v = Wl[i - OFF_WL];
    else                   v = Wr[i - OFF_WR];
    wh[i] = (_Float16)v;
}

__global__ void gat_init(const float* __restrict__ bias_out, float* __restrict__ z_out,
                         unsigned* __restrict__ m_enc, float* __restrict__ sbuf) {
    size_t i = (size_t)blockIdx.x * blockDim.x + threadIdx.x;
    if (i < (size_t)N_NODES * G_F) z_out[i] = bias_out[i & (G_F - 1)];
    if (i < N_NODES) { m_enc[i] = ENC_NEG_INF; sbuf[i] = 0.f; }
}

// Fused: MLP encoder (2 layers) -> GRU cell -> GATv2 linear transforms.
// One wave handles 16 node rows; 4 waves per block; all GEMMs via v_wmma_f32_16x16x32_f16.
__global__ void __launch_bounds__(128)
fused_dense(const float* __restrict__ in, const float* __restrict__ hid,
            const _Float16* __restrict__ wh,
            const float* __restrict__ b0, const float* __restrict__ b1,
            const float* __restrict__ b_ih, const float* __restrict__ b_hh,
            const float* __restrict__ bl, const float* __restrict__ br,
            float* __restrict__ h_out, float* __restrict__ xl, float* __restrict__ xr) {
    __shared__ _Float16 smem[4][3][16 * 128];   // per-wave: X1/h', X2, h_in(f16)
    const int wid  = threadIdx.x >> 5;
    const int lane = threadIdx.x & 31;
    const int wave = blockIdx.x * 4 + wid;
    if (wave >= N_NODES / 16) return;
    const int nb   = wave * 16;
    const int m    = lane & 15;     // A row / B,D column index
    const int half = lane >> 4;

    _Float16* bufA = smem[wid][0];  // X1, later h'
    _Float16* bufB = smem[wid][1];  // X2
    _Float16* bufH = smem[wid][2];  // h_in as f16

    // stage h_in -> LDS f16
    {
        const float* hb = hid + (size_t)nb * M_F;
        for (int i = lane; i < 16 * M_F; i += 32) bufH[i] = (_Float16)hb[i];
    }

    // ---- stage 0: X1 = relu(in @ W0^T + b0)   [16 x 128], K = 96
    {
        const float* xrow = in + (size_t)(nb + m) * IN_F;
        v16h a0 = loadA_g(xrow, 0, half);
        v16h a1 = loadA_g(xrow, 32, half);
        v16h a2 = loadA_g(xrow, 64, half);
        const _Float16* W0h = wh + OFF_W0;
#pragma unroll
        for (int t = 0; t < 8; ++t) {
            v8f acc = {};
            acc = wmma_f16(a0, loadB(W0h, IN_F, t * 16, 0, m, half), acc);
            acc = wmma_f16(a1, loadB(W0h, IN_F, t * 16, 32, m, half), acc);
            acc = wmma_f16(a2, loadB(W0h, IN_F, t * 16, 64, m, half), acc);
            float bv = b0[t * 16 + m];
#pragma unroll
            for (int r = 0; r < 8; ++r) {
                float v = acc[r] + bv;
                v = v > 0.f ? v : 0.f;
                bufA[(r + half * 8) * H_F + t * 16 + m] = (_Float16)v;
            }
        }
    }

    // ---- stage 1: X2 = relu(X1 @ W1^T + b1)   [16 x 128], K = 128
    {
        const _Float16* W1h = wh + OFF_W1;
#pragma unroll
        for (int t = 0; t < 8; ++t) {
            v8f acc = {};
#pragma unroll
            for (int ks = 0; ks < 4; ++ks) {
                v16h a = loadA_lds(bufA, H_F, ks * 32, m, half);
                acc = wmma_f16(a, loadB(W1h, H_F, t * 16, ks * 32, m, half), acc);
            }
            float bv = b1[t * 16 + m];
#pragma unroll
            for (int r = 0; r < 8; ++r) {
                float v = acc[r] + bv;
                v = v > 0.f ? v : 0.f;
                bufB[(r + half * 8) * H_F + t * 16 + m] = (_Float16)v;
            }
        }
    }

    // ---- stage 2: GRU cell, h' written to h_out (f32) and bufA (f16)
    {
        const _Float16* Wihh = wh + OFF_WIH;
        const _Float16* Whhh = wh + OFF_WHH;
#pragma unroll 1
        for (int t = 0; t < 8; ++t) {
            v8f ir = {}, iz = {}, in2 = {}, hr = {}, hz = {}, hn = {};
#pragma unroll
            for (int ks = 0; ks < 4; ++ks) {
                v16h ax = loadA_lds(bufB, M_F, ks * 32, m, half);
                v16h ah = loadA_lds(bufH, M_F, ks * 32, m, half);
                ir  = wmma_f16(ax, loadB(Wihh, M_F,       t * 16, ks * 32, m, half), ir);
                iz  = wmma_f16(ax, loadB(Wihh, M_F, 128 + t * 16, ks * 32, m, half), iz);
                in2 = wmma_f16(ax, loadB(Wihh, M_F, 256 + t * 16, ks * 32, m, half), in2);
                hr  = wmma_f16(ah, loadB(Whhh, M_F,       t * 16, ks * 32, m, half), hr);
                hz  = wmma_f16(ah, loadB(Whhh, M_F, 128 + t * 16, ks * 32, m, half), hz);
                hn  = wmma_f16(ah, loadB(Whhh, M_F, 256 + t * 16, ks * 32, m, half), hn);
            }
            float bir = b_ih[t * 16 + m], biz = b_ih[128 + t * 16 + m], bin_ = b_ih[256 + t * 16 + m];
            float bhr = b_hh[t * 16 + m], bhz = b_hh[128 + t * 16 + m], bhn  = b_hh[256 + t * 16 + m];
#pragma unroll
            for (int r = 0; r < 8; ++r) {
                size_t row = (size_t)(nb + r + half * 8);
                float hvv = hid[row * M_F + t * 16 + m];
                float rg = sigmoidf_((ir[r] + bir) + (hr[r] + bhr));
                float zg = sigmoidf_((iz[r] + biz) + (hz[r] + bhz));
                float ng = tanhf_((in2[r] + bin_) + rg * (hn[r] + bhn));
                float hnew = (1.f - zg) * ng + zg * hvv;
                h_out[row * M_F + t * 16 + m] = hnew;
                bufA[(r + half * 8) * M_F + t * 16 + m] = (_Float16)hnew;
            }
        }
    }

    // ---- stage 3: xl = h' @ Wl^T + bl ; xr = h' @ Wr^T + br   [16 x 64]
    {
        const _Float16* Wlh = wh + OFF_WL;
        const _Float16* Wrh = wh + OFF_WR;
#pragma unroll
        for (int t = 0; t < 4; ++t) {
            v8f al = {}, ar = {};
#pragma unroll
            for (int ks = 0; ks < 4; ++ks) {
                v16h a = loadA_lds(bufA, M_F, ks * 32, m, half);
                al = wmma_f16(a, loadB(Wlh, M_F, t * 16, ks * 32, m, half), al);
                ar = wmma_f16(a, loadB(Wrh, M_F, t * 16, ks * 32, m, half), ar);
            }
            float blv = bl[t * 16 + m], brv = br[t * 16 + m];
#pragma unroll
            for (int r = 0; r < 8; ++r) {
                size_t row = (size_t)(nb + r + half * 8);
                xl[row * G_F + t * 16 + m] = al[r] + blv;
                xr[row * G_F + t * 16 + m] = ar[r] + brv;
            }
        }
    }
}

// pass 1: per-edge attention logit e = att . leaky_relu(xl[src] + xr[dst]); segment max via atomic
__global__ void gat_logits(const int* __restrict__ ei, const float* __restrict__ xl,
                           const float* __restrict__ xr, const float* __restrict__ att,
                           float* __restrict__ ebuf, unsigned* __restrict__ m_enc) {
    int t = blockIdx.x * blockDim.x + threadIdx.x;
    int e = t >> 4;
    if (e >= N_EDGES) return;
    int fl = t & 15;
    int s = ei[e], d = ei[N_EDGES + e];
    v4f a  = *(const v4f*)(xl + (size_t)s * G_F + fl * 4);
    v4f b  = *(const v4f*)(xr + (size_t)d * G_F + fl * 4);
    v4f av = *(const v4f*)(att + fl * 4);
    float acc = 0.f;
#pragma unroll
    for (int i = 0; i < 4; ++i) {
        float v = a[i] + b[i];
        v = v > 0.f ? v : 0.2f * v;
        acc += v * av[i];
    }
    acc += __shfl_xor(acc, 1);
    acc += __shfl_xor(acc, 2);
    acc += __shfl_xor(acc, 4);
    acc += __shfl_xor(acc, 8);
    if (fl == 0) {
        ebuf[e] = acc;
        atomicMax(&m_enc[d], enc_f32(acc));
    }
}

// pass 2: p = exp(e - m[dst]); segment sum via f32 atomic add (L2 atomics)
__global__ void gat_softmax_num(const int* __restrict__ ei, float* __restrict__ ep,
                                const unsigned* __restrict__ m_enc, float* __restrict__ sbuf) {
    int e = blockIdx.x * blockDim.x + threadIdx.x;
    if (e >= N_EDGES) return;
    int d = ei[N_EDGES + e];
    float p = __expf(ep[e] - dec_f32(m_enc[d]));
    ep[e] = p;
    fadd_agent(&sbuf[d], p);
}

// pass 3: z[dst] += (p / s[dst]) * xl[src]
__global__ void gat_aggregate(const int* __restrict__ ei, const float* __restrict__ xl,
                              const float* __restrict__ pbuf, const float* __restrict__ sbuf,
                              float* __restrict__ z_out) {
    int t = blockIdx.x * blockDim.x + threadIdx.x;
    int e = t >> 4;
    if (e >= N_EDGES) return;
    int fl = t & 15;
    int s = ei[e], d = ei[N_EDGES + e];
    float sv = sbuf[d];
    float alpha = pbuf[e] / (sv > 0.f ? sv : 1.f);
    v4f a = *(const v4f*)(xl + (size_t)s * G_F + fl * 4);
#pragma unroll
    for (int i = 0; i < 4; ++i)
        fadd_agent(&z_out[(size_t)d * G_F + fl * 4 + i], alpha * a[i]);
}

// ---------------------------------------------------------------- launch

extern "C" void kernel_launch(void* const* d_in, const int* in_sizes, int n_in,
                              void* d_out, int out_size, void* d_ws, size_t ws_size,
                              hipStream_t stream) {
    (void)in_sizes; (void)n_in; (void)out_size; (void)ws_size;
    const float* in     = (const float*)d_in[0];
    const float* hid    = (const float*)d_in[1];
    const int*   ei     = (const int*)d_in[2];
    const float* W0     = (const float*)d_in[3];
    const float* b0     = (const float*)d_in[4];
    const float* W1     = (const float*)d_in[5];
    const float* b1     = (const float*)d_in[6];
    const float* W_ih   = (const float*)d_in[7];
    const float* b_ih   = (const float*)d_in[8];
    const float* W_hh   = (const float*)d_in[9];
    const float* b_hh   = (const float*)d_in[10];
    const float* Wl     = (const float*)d_in[11];
    const float* bl     = (const float*)d_in[12];
    const float* Wr     = (const float*)d_in[13];
    const float* br     = (const float*)d_in[14];
    const float* att    = (const float*)d_in[15];
    const float* b_out  = (const float*)d_in[16];

    float* z_out = (float*)d_out;                       // [N, 64]
    float* h_out = z_out + (size_t)N_NODES * G_F;       // [N, 128]

    _Float16* wh = (_Float16*)d_ws;                     // f16 weights
    float* wsf   = (float*)d_ws;
    float* xl    = wsf + WH_TOTAL / 2;                  // [N, 64]
    float* xr    = xl + (size_t)N_NODES * G_F;          // [N, 64]
    float* ep    = xr + (size_t)N_NODES * G_F;          // [E] logits -> p
    unsigned* m_enc = (unsigned*)(ep + N_EDGES);        // [N]
    float* sbuf  = (float*)(m_enc + N_NODES);           // [N]

    cvt_weights<<<(WH_TOTAL + 255) / 256, 256, 0, stream>>>(W0, W1, W_ih, W_hh, Wl, Wr, wh);
    gat_init<<<((size_t)N_NODES * G_F + 255) / 256, 256, 0, stream>>>(b_out, z_out, m_enc, sbuf);
    fused_dense<<<(N_NODES / 16 + 3) / 4, 128, 0, stream>>>(in, hid, wh, b0, b1, b_ih, b_hh,
                                                            bl, br, h_out, xl, xr);
    gat_logits<<<((size_t)N_EDGES * 16 + 255) / 256, 256, 0, stream>>>(ei, xl, xr, att, ep, m_enc);
    gat_softmax_num<<<(N_EDGES + 255) / 256, 256, 0, stream>>>(ei, ep, m_enc, sbuf);
    gat_aggregate<<<((size_t)N_EDGES * 16 + 255) / 256, 256, 0, stream>>>(ei, xl, ep, sbuf, z_out);
}
